// AttentionBlock_55499567399250
// MI455X (gfx1250) — matrix operand order repaired and checked by reading, not compiled, not run on hardware
//
#include <hip/hip_runtime.h>

#define SEQ 2048
#define HID 2048
#define HEADS 16
#define DH 128

typedef __attribute__((ext_vector_type(16))) __bf16 v16bf;
typedef __attribute__((ext_vector_type(8)))  float  v8f;

union Frag {
    v16bf v;
    uint4 q[2];
    unsigned short h[16];
};

__device__ __forceinline__ unsigned short f2bf(float f) {
    unsigned int u = __float_as_uint(f);
    unsigned int r = 0x7fffu + ((u >> 16) & 1u);
    return (unsigned short)((u + r) >> 16);
}

// ---------------------------------------------------------------------------
// 0) f32 -> bf16 conversion
// ---------------------------------------------------------------------------
__global__ __launch_bounds__(256) void conv_bf16_kernel(const float* __restrict__ src,
                                                        unsigned short* __restrict__ dst,
                                                        int n) {
    int i = blockIdx.x * blockDim.x + threadIdx.x;
    if (i < n) dst[i] = f2bf(src[i]);
}

// ---------------------------------------------------------------------------
// 1) QKV projection + bias + RoPE (Q,K) + transposed store of V.
//    grid = (HID/128 heads, SEQ/64, 3), block = 128 (4 waves).
//    Wave w computes rows [mtile*64 + w*16, +16) x one full head (128 cols).
// ---------------------------------------------------------------------------
__global__ __launch_bounds__(128) void qkv_rope_kernel(
    const unsigned short* __restrict__ xb,
    const unsigned short* __restrict__ Wqb,
    const unsigned short* __restrict__ Wkb,
    const unsigned short* __restrict__ Wvb,
    const float* __restrict__ bq,
    const float* __restrict__ bk,
    const float* __restrict__ bv,
    unsigned short* __restrict__ Qb,
    unsigned short* __restrict__ Kb,
    unsigned short* __restrict__ Vtb) {

    const int z = blockIdx.z;                       // 0=Q 1=K 2=V
    const unsigned short* W = (z == 0) ? Wqb : (z == 1) ? Wkb : Wvb;
    const float* bias       = (z == 0) ? bq  : (z == 1) ? bk  : bv;

    const int tid  = threadIdx.x;
    const int w    = tid >> 5;
    const int l    = tid & 31;
    const int ln16 = l & 15;
    const int half = (l >> 4) & 1;

    const int head    = blockIdx.x;
    const int rowbase = blockIdx.y * 64 + w * 16;
    const int colbase = head * 128;

    v8f acc[8] = {};

    const unsigned short* arow = xb + (size_t)(rowbase + ln16) * HID;

    for (int k = 0; k < HID; k += 32) {
        Frag a;
        const unsigned short* pa = arow + k + half * 8;
        a.q[0] = *reinterpret_cast<const uint4*>(pa);
        a.q[1] = *reinterpret_cast<const uint4*>(pa + 16);
#pragma unroll
        for (int nt = 0; nt < 8; ++nt) {
            const int n = colbase + nt * 16 + ln16;
            const unsigned short* pb = W + (size_t)n * HID + k + half * 8;
            Frag b;
            b.q[0] = *reinterpret_cast<const uint4*>(pb);
            b.q[1] = *reinterpret_cast<const uint4*>(pb + 16);
            acc[nt] = __builtin_amdgcn_wmma_f32_16x16x32_bf16(
                false, a.v, false, b.v, (short)0, acc[nt], false, false);
        }
    }

    // bias
#pragma unroll
    for (int nt = 0; nt < 8; ++nt) {
        const float bi = bias[colbase + nt * 16 + ln16];
#pragma unroll
        for (int v = 0; v < 8; ++v) acc[nt][v] += bi;
    }

    if (z < 2) {
        // RoPE: within one head, pairs (d, d+64) sit in acc[nt] / acc[nt+4],
        // same lane, same slot -> pure per-lane rotation.
#pragma unroll
        for (int nt = 0; nt < 4; ++nt) {
            const int i = nt * 16 + ln16;                 // d % 64
            const float invf = exp2f(-(float)i * 0.20762050593046014f); // theta^(-2i/128)
#pragma unroll
            for (int v = 0; v < 8; ++v) {
                const float pos = (float)(rowbase + v + half * 8);
                const float ang = pos * invf;
                const float c = __cosf(ang), s = __sinf(ang);
                const float x1 = acc[nt][v];
                const float x2 = acc[nt + 4][v];
                acc[nt][v]     = x1 * c - x2 * s;
                acc[nt + 4][v] = x2 * c + x1 * s;
            }
        }
        unsigned short* dst = (z == 0) ? Qb : Kb;
#pragma unroll
        for (int nt = 0; nt < 8; ++nt) {
            const int n = colbase + nt * 16 + ln16;
#pragma unroll
            for (int v = 0; v < 8; ++v) {
                const int row = rowbase + v + half * 8;
                dst[(size_t)row * HID + n] = f2bf(acc[nt][v]);
            }
        }
    } else {
        // V stored transposed: Vt[d][s]; each lane writes 8 contiguous bf16.
#pragma unroll
        for (int nt = 0; nt < 8; ++nt) {
            const int n = colbase + nt * 16 + ln16;
            union { unsigned short h[8]; uint4 q; } pk;
#pragma unroll
            for (int v = 0; v < 8; ++v) pk.h[v] = f2bf(acc[nt][v]);
            *reinterpret_cast<uint4*>(Vtb + (size_t)n * SEQ + rowbase + half * 8) = pk.q;
        }
    }
}

// ---------------------------------------------------------------------------
// 2) Causal flash attention. grid = (SEQ/64, HEADS), block = 128 (4 waves).
//    Wave w owns 16 query rows; online softmax; P staged through LDS to
//    convert C-layout -> A-layout for the P*V WMMA.
// ---------------------------------------------------------------------------
__global__ __launch_bounds__(128) void attn_kernel(
    const unsigned short* __restrict__ Qb,
    const unsigned short* __restrict__ Kb,
    const unsigned short* __restrict__ Vtb,
    unsigned short* __restrict__ Ab) {

    __shared__ __align__(16) unsigned short Pls[4 * 16 * 64];

    const int tid  = threadIdx.x;
    const int w    = tid >> 5;
    const int l    = tid & 31;
    const int ln16 = l & 15;
    const int half = (l >> 4) & 1;

    const int qb = blockIdx.x;
    const int h  = blockIdx.y;
    const int qrowbase = qb * 64 + w * 16;
    const int lbase = w * 16 * 64;

    // Q fragments for the full head dim (K = 128 in 4 chunks of 32)
    Frag qf[4];
#pragma unroll
    for (int kk = 0; kk < 4; ++kk) {
        const unsigned short* pa =
            Qb + (size_t)(qrowbase + ln16) * HID + h * 128 + kk * 32 + half * 8;
        qf[kk].q[0] = *reinterpret_cast<const uint4*>(pa);
        qf[kk].q[1] = *reinterpret_cast<const uint4*>(pa + 16);
    }

    float rmax[8], rsum[8];
#pragma unroll
    for (int v = 0; v < 8; ++v) { rmax[v] = -1e30f; rsum[v] = 0.0f; }
    v8f oacc[8] = {};

    const float sc = 0.08838834764831845f; // 1/sqrt(128)

    for (int kb = 0; kb <= qb; ++kb) {
        const int knmax = (kb == qb) ? w : 3;

        // S = Q * K^T strip (16 x 64), 4 key sub-tiles
        v8f sacc[4] = {};
        for (int kn = 0; kn <= knmax; ++kn) {
            const int key = kb * 64 + kn * 16 + ln16;
#pragma unroll
            for (int kk = 0; kk < 4; ++kk) {
                const unsigned short* pb =
                    Kb + (size_t)key * HID + h * 128 + kk * 32 + half * 8;
                Frag b;
                b.q[0] = *reinterpret_cast<const uint4*>(pb);
                b.q[1] = *reinterpret_cast<const uint4*>(pb + 16);
                sacc[kn] = __builtin_amdgcn_wmma_f32_16x16x32_bf16(
                    false, qf[kk].v, false, b.v, (short)0, sacc[kn], false, false);
            }
        }

        // scale + causal mask (diagonal tile only)
        for (int kn = 0; kn <= knmax; ++kn) {
#pragma unroll
            for (int v = 0; v < 8; ++v) {
                float s = sacc[kn][v] * sc;
                if (kb == qb && kn == w)
                    s = (ln16 > v + half * 8) ? -1e30f : s;
                sacc[kn][v] = s;
            }
        }

        // per-row block max (reduce across 16 lanes of the half-wave)
        float bm[8];
#pragma unroll
        for (int v = 0; v < 8; ++v) {
            float m = sacc[0][v];
            for (int kn = 1; kn <= knmax; ++kn) m = fmaxf(m, sacc[kn][v]);
            m = fmaxf(m, __shfl_xor(m, 8, 16));
            m = fmaxf(m, __shfl_xor(m, 4, 16));
            m = fmaxf(m, __shfl_xor(m, 2, 16));
            m = fmaxf(m, __shfl_xor(m, 1, 16));
            bm[v] = m;
        }

        float alpha[8];
#pragma unroll
        for (int v = 0; v < 8; ++v) {
            const float nm = fmaxf(rmax[v], bm[v]);
            alpha[v] = __expf(rmax[v] - nm);
            rmax[v] = nm;
        }

        // P = exp(s - m), row sums
#pragma unroll
        for (int v = 0; v < 8; ++v) {
            float part = 0.0f;
            for (int kn = 0; kn <= knmax; ++kn) {
                const float p = __expf(sacc[kn][v] - rmax[v]);
                sacc[kn][v] = p;
                part += p;
            }
            part += __shfl_xor(part, 8, 16);
            part += __shfl_xor(part, 4, 16);
            part += __shfl_xor(part, 2, 16);
            part += __shfl_xor(part, 1, 16);
            rsum[v] = rsum[v] * alpha[v] + part;
        }

        // rescale running output
#pragma unroll
        for (int dt = 0; dt < 8; ++dt)
#pragma unroll
            for (int v = 0; v < 8; ++v) oacc[dt][v] *= alpha[v];

        // stage P (bf16) through LDS: C-layout -> row-major [16][64]
#pragma unroll
        for (int kn = 0; kn < 4; ++kn) {
#pragma unroll
            for (int v = 0; v < 8; ++v) {
                unsigned short hv = 0;
                if (kn <= knmax) hv = f2bf(sacc[kn][v]);
                Pls[lbase + (v + half * 8) * 64 + kn * 16 + ln16] = hv;
            }
        }

        // re-fragment P as A-layout (K = 64 keys in 2 chunks of 32)
        Frag pf[2];
#pragma unroll
        for (int c = 0; c < 2; ++c) {
            const unsigned short* pp = &Pls[lbase + ln16 * 64 + c * 32 + half * 8];
            pf[c].q[0] = *reinterpret_cast<const uint4*>(pp);
            pf[c].q[1] = *reinterpret_cast<const uint4*>(pp + 16);
        }

        // O += P * V  (V^T rows are contiguous in keys)
#pragma unroll
        for (int dt = 0; dt < 8; ++dt) {
            const int d = h * 128 + dt * 16 + ln16;
#pragma unroll
            for (int c = 0; c < 2; ++c) {
                const unsigned short* pv =
                    Vtb + (size_t)d * SEQ + kb * 64 + c * 32 + half * 8;
                Frag b;
                b.q[0] = *reinterpret_cast<const uint4*>(pv);
                b.q[1] = *reinterpret_cast<const uint4*>(pv + 16);
                oacc[dt] = __builtin_amdgcn_wmma_f32_16x16x32_bf16(
                    false, pf[c].v, false, b.v, (short)0, oacc[dt], false, false);
            }
        }
    }

    // normalize and write context (bf16, [S][HID])
    float inv[8];
#pragma unroll
    for (int v = 0; v < 8; ++v) inv[v] = 1.0f / rsum[v];
#pragma unroll
    for (int dt = 0; dt < 8; ++dt) {
        const int n = h * 128 + dt * 16 + ln16;
#pragma unroll
        for (int v = 0; v < 8; ++v) {
            const int row = qrowbase + v + half * 8;
            Ab[(size_t)row * HID + n] = f2bf(oacc[dt][v] * inv[v]);
        }
    }
}

// ---------------------------------------------------------------------------
// 3) Output projection + bias + residual. grid = (HID/128, SEQ/64), block 128.
// ---------------------------------------------------------------------------
__global__ __launch_bounds__(128) void proj_res_kernel(
    const unsigned short* __restrict__ Ab,
    const unsigned short* __restrict__ Wob,
    const float* __restrict__ bo,
    const float* __restrict__ x,
    float* __restrict__ Res) {

    const int tid  = threadIdx.x;
    const int w    = tid >> 5;
    const int l    = tid & 31;
    const int ln16 = l & 15;
    const int half = (l >> 4) & 1;

    const int colbase = blockIdx.x * 128;
    const int rowbase = blockIdx.y * 64 + w * 16;

    v8f acc[8] = {};
    const unsigned short* arow = Ab + (size_t)(rowbase + ln16) * HID;

    for (int k = 0; k < HID; k += 32) {
        Frag a;
        const unsigned short* pa = arow + k + half * 8;
        a.q[0] = *reinterpret_cast<const uint4*>(pa);
        a.q[1] = *reinterpret_cast<const uint4*>(pa + 16);
#pragma unroll
        for (int nt = 0; nt < 8; ++nt) {
            const int n = colbase + nt * 16 + ln16;
            const unsigned short* pb = Wob + (size_t)n * HID + k + half * 8;
            Frag b;
            b.q[0] = *reinterpret_cast<const uint4*>(pb);
            b.q[1] = *reinterpret_cast<const uint4*>(pb + 16);
            acc[nt] = __builtin_amdgcn_wmma_f32_16x16x32_bf16(
                false, a.v, false, b.v, (short)0, acc[nt], false, false);
        }
    }

#pragma unroll
    for (int nt = 0; nt < 8; ++nt) {
        const int n = colbase + nt * 16 + ln16;
        const float bi = bo[n];
#pragma unroll
        for (int v = 0; v < 8; ++v) {
            const int row = rowbase + v + half * 8;
            Res[(size_t)row * HID + n] = acc[nt][v] + bi + x[(size_t)row * HID + n];
        }
    }
}

// ---------------------------------------------------------------------------
// 4) Row LayerNorm. grid = SEQ, block = 256.
// ---------------------------------------------------------------------------
__global__ __launch_bounds__(256) void ln_kernel(
    const float* __restrict__ Res,
    const float* __restrict__ ln_w,
    const float* __restrict__ ln_b,
    float* __restrict__ out) {

    __shared__ float s1[256];
    __shared__ float s2[256];
    const int r = blockIdx.x;
    const int t = threadIdx.x;
    const float* row = Res + (size_t)r * HID;

    float vals[8];
    float sum = 0.0f, sq = 0.0f;
#pragma unroll
    for (int i = 0; i < 8; ++i) {
        const float v = row[i * 256 + t];
        vals[i] = v;
        sum += v;
        sq += v * v;
    }
    s1[t] = sum; s2[t] = sq;
    __syncthreads();
    for (int st = 128; st > 0; st >>= 1) {
        if (t < st) { s1[t] += s1[t + st]; s2[t] += s2[t + st]; }
        __syncthreads();
    }
    const float mean = s1[0] * (1.0f / HID);
    const float var  = s2[0] * (1.0f / HID) - mean * mean;
    const float inv  = rsqrtf(var + 1e-5f);
#pragma unroll
    for (int i = 0; i < 8; ++i) {
        const int c = i * 256 + t;
        out[(size_t)r * HID + c] = (vals[i] - mean) * inv * ln_w[c] + ln_b[c];
    }
}

// ---------------------------------------------------------------------------
extern "C" void kernel_launch(void* const* d_in, const int* in_sizes, int n_in,
                              void* d_out, int out_size, void* d_ws, size_t ws_size,
                              hipStream_t stream) {
    const float* x    = (const float*)d_in[0];
    const float* Wq   = (const float*)d_in[1];
    const float* bq   = (const float*)d_in[2];
    const float* Wk   = (const float*)d_in[3];
    const float* bk   = (const float*)d_in[4];
    const float* Wv   = (const float*)d_in[5];
    const float* bv   = (const float*)d_in[6];
    const float* Wo   = (const float*)d_in[7];
    const float* bo   = (const float*)d_in[8];
    const float* ln_w = (const float*)d_in[9];
    const float* ln_b = (const float*)d_in[10];

    char* wsb = (char*)d_ws;
    const size_t MB8 = (size_t)SEQ * HID * sizeof(unsigned short); // 8 MiB
    unsigned short* xb  = (unsigned short*)(wsb + 0 * MB8);
    unsigned short* Wqb = (unsigned short*)(wsb + 1 * MB8);
    unsigned short* Wkb = (unsigned short*)(wsb + 2 * MB8);
    unsigned short* Wvb = (unsigned short*)(wsb + 3 * MB8);
    unsigned short* Wob = (unsigned short*)(wsb + 4 * MB8);
    unsigned short* Qb  = (unsigned short*)(wsb + 5 * MB8);
    unsigned short* Kb  = (unsigned short*)(wsb + 6 * MB8);
    unsigned short* Vtb = (unsigned short*)(wsb + 7 * MB8);
    unsigned short* Ab  = (unsigned short*)(wsb + 8 * MB8);
    float*          Res = (float*)(wsb + 9 * MB8);

    const int nElem = SEQ * HID;
    const int cgrid = (nElem + 255) / 256;
    conv_bf16_kernel<<<cgrid, 256, 0, stream>>>(x,  xb,  nElem);
    conv_bf16_kernel<<<cgrid, 256, 0, stream>>>(Wq, Wqb, nElem);
    conv_bf16_kernel<<<cgrid, 256, 0, stream>>>(Wk, Wkb, nElem);
    conv_bf16_kernel<<<cgrid, 256, 0, stream>>>(Wv, Wvb, nElem);
    conv_bf16_kernel<<<cgrid, 256, 0, stream>>>(Wo, Wob, nElem);

    qkv_rope_kernel<<<dim3(HID / 128, SEQ / 64, 3), 128, 0, stream>>>(
        xb, Wqb, Wkb, Wvb, bq, bk, bv, Qb, Kb, Vtb);

    attn_kernel<<<dim3(SEQ / 64, HEADS), 128, 0, stream>>>(Qb, Kb, Vtb, Ab);

    proj_res_kernel<<<dim3(HID / 128, SEQ / 64), 128, 0, stream>>>(Ab, Wob, bo, x, Res);

    ln_kernel<<<SEQ, 256, 0, stream>>>(Res, ln_w, ln_b, (float*)d_out);
}
